// pointnet2_ssg_62388694942182
// MI455X (gfx1250) — matrix operand order repaired and checked by reading, not compiled, not run on hardware
//
#include <hip/hip_runtime.h>
#include <hip/hip_bf16.h>
#include <stdint.h>

// ---------------------------------------------------------------------------
// PointNet++ SSG forward for gfx1250 (MI455X).
// MLP layers = WMMA f32<-f16 GEMMs (v_wmma_f32_16x16x32_f16), guard-free inner
// loop (K pre-padded to 32, weights pre-converted to f16), BN+bias folded into
// scale/shift epilogue with fused ReLU. Two 16x16 tiles per wave.
// ---------------------------------------------------------------------------

#define BATCH 16
#define NPTS  4096
#define CINR  9
#define EPSBN 1e-5f

typedef __attribute__((ext_vector_type(16))) _Float16 v16h;
typedef __attribute__((ext_vector_type(8)))  _Float16 v8h;
typedef __attribute__((ext_vector_type(8)))  float    v8f;

__device__ __forceinline__ void cvt4(v16h& v, int base, float4 f) {
    v[base + 0] = (_Float16)f.x; v[base + 1] = (_Float16)f.y;
    v[base + 2] = (_Float16)f.z; v[base + 3] = (_Float16)f.w;
}

// ----------------------------- WMMA GEMM -----------------------------------
// Y[co, col] = relu( (sum_k Wh[co,k] * X[k,col]) * s[co] + t[co] )
// Wh: f16, row stride Kp (Kp % 32 == 0, zero padded). X: f32, Kp rows valid.
// Block = 256 thr = 8 waves -> 32(co) x 128(col); wave -> 16(co) x 32(col).
__global__ __launch_bounds__(256)
void gemm_bn_relu(const _Float16* __restrict__ Wh, const float* __restrict__ X,
                  float* __restrict__ Y, int Co, int Kp, int Cols,
                  const float* __restrict__ bias, const float* __restrict__ gam,
                  const float* __restrict__ bet, const float* __restrict__ mu,
                  const float* __restrict__ var, int do_relu)
{
    const int lane = threadIdx.x & 31;
    const int wave = threadIdx.x >> 5;          // 0..7
    const int wm   = wave >> 2;                 // 0..1
    const int wn   = wave & 3;                  // 0..3
    const int coBase  = blockIdx.y * 32 + wm * 16;
    const int colBase = blockIdx.x * 128 + wn * 32;
    const int lm = lane & 15;
    const int lg = lane >> 4;                   // 0/1

    v8f acc0 = {0.f,0.f,0.f,0.f,0.f,0.f,0.f,0.f};
    v8f acc1 = {0.f,0.f,0.f,0.f,0.f,0.f,0.f,0.f};

    // A: lane holds row m = coBase+lm; halves 0..7 at k=kc+lg*8, 8..15 at +16.
    const _Float16* wrow = Wh + (size_t)(coBase + lm) * Kp + lg * 8;
    // B: lane holds col n = colBase+lm; halves 0..7 at k=kc+lg*8, 8..15 at +16.
    const float* xbase = X + (size_t)(lg * 8) * Cols + colBase + lm;

    for (int kc = 0; kc < Kp; kc += 32) {
        v8h a0 = *(const v8h*)(wrow + kc);
        v8h a1 = *(const v8h*)(wrow + kc + 16);
        v16h a = __builtin_shufflevector(a0, a1, 0,1,2,3,4,5,6,7,
                                                8,9,10,11,12,13,14,15);
        const float* xp = xbase + (size_t)kc * Cols;
        v16h bf, bg;
        #pragma unroll
        for (int s = 0; s < 16; ++s) {
            size_t o = (size_t)(s < 8 ? s : s + 8) * Cols;
            bf[s] = (_Float16)xp[o]; bg[s] = (_Float16)xp[o + 16];
        }
        acc0 = __builtin_amdgcn_wmma_f32_16x16x32_f16(false, a, false, bf,
                                                      (short)0, acc0, false, false);
        acc1 = __builtin_amdgcn_wmma_f32_16x16x32_f16(false, a, false, bg,
                                                      (short)0, acc1, false, false);
    }

    // C/D layout: lane (l&15) = N, VGPR r -> M = r + 8*(l>>4)
    const int ncol0 = colBase + lm;
    #pragma unroll
    for (int r = 0; r < 8; ++r) {
        int m  = coBase + r + lg * 8;
        float s = gam[m] * rsqrtf(var[m] + EPSBN);
        float t = (bias[m] - mu[m]) * s + bet[m];
        float y0 = acc0[r] * s + t;
        float y1 = acc1[r] * s + t;
        if (do_relu) { y0 = fmaxf(y0, 0.0f); y1 = fmaxf(y1, 0.0f); }
        Y[(size_t)m * Cols + ncol0]      = y0;
        Y[(size_t)m * Cols + ncol0 + 16] = y1;
    }
}

// -------------------- weight f32 -> f16 with K padding -----------------------
__global__ void wcvt_kernel(const float* __restrict__ W, _Float16* __restrict__ Wh,
                            int Co, int K, int Kp)
{
    int t = blockIdx.x * blockDim.x + threadIdx.x;
    if (t >= Co * Kp) return;
    int co = t / Kp, k = t - co * Kp;
    Wh[t] = (k < K) ? (_Float16)W[(size_t)co * K + k] : (_Float16)0.0f;
}

// ----------------------------- zero fill -------------------------------------
__global__ void fill0_kernel(float* __restrict__ p, long long n)
{
    long long t = (long long)blockIdx.x * blockDim.x + threadIdx.x;
    if (t < n) p[t] = 0.0f;
}

// ----------------------------- FPS ------------------------------------------
// One block per batch; dist[] in LDS; argmax with first-index tiebreak.
// base = xyz + b*bstride; element = base[c*cstride + n]
__global__ __launch_bounds__(256)
void fps_kernel(const float* __restrict__ xyz, int bstride, int cstride,
                int n, int npoint, int* __restrict__ idx_out,
                float* __restrict__ nxyz, int Bnp)
{
    __shared__ float dist[4096];
    __shared__ float rv[256];
    __shared__ int   ri[256];
    const int b = blockIdx.x;
    const float* base = xyz + (size_t)b * bstride;
    for (int i = threadIdx.x; i < n; i += 256) dist[i] = 1e10f;
    __syncthreads();

    int far = 0;
    for (int it = 0; it < npoint; ++it) {
        float cx = base[0 * cstride + far];
        float cy = base[1 * cstride + far];
        float cz = base[2 * cstride + far];
        if (threadIdx.x == 0) {
            idx_out[b * npoint + it] = far;
            nxyz[0 * Bnp + b * npoint + it] = cx;
            nxyz[1 * Bnp + b * npoint + it] = cy;
            nxyz[2 * Bnp + b * npoint + it] = cz;
        }
        float best = -1.0f; int bi = n;
        for (int i = threadIdx.x; i < n; i += 256) {
            float dx = base[0 * cstride + i] - cx;
            float dy = base[1 * cstride + i] - cy;
            float dz = base[2 * cstride + i] - cz;
            float d  = dx * dx + dy * dy + dz * dz;
            float dm = fminf(dist[i], d);
            dist[i] = dm;
            if (dm > best) { best = dm; bi = i; }
        }
        rv[threadIdx.x] = best; ri[threadIdx.x] = bi;
        __syncthreads();
        for (int s = 128; s > 0; s >>= 1) {
            if (threadIdx.x < s) {
                float ov = rv[threadIdx.x + s]; int oi = ri[threadIdx.x + s];
                if (ov > rv[threadIdx.x] ||
                    (ov == rv[threadIdx.x] && oi < ri[threadIdx.x])) {
                    rv[threadIdx.x] = ov; ri[threadIdx.x] = oi;
                }
            }
            __syncthreads();
        }
        far = ri[0];
        __syncthreads();
    }
}

// ----------------------------- ball query -----------------------------------
__global__ void query_ball_kernel(const float* __restrict__ xyz, int bstride,
                                  int cstride, int n,
                                  const float* __restrict__ nxyz, int Bnp,
                                  int npoint, int nsample, float r2,
                                  int* __restrict__ gidx)
{
    int t = blockIdx.x * blockDim.x + threadIdx.x;
    if (t >= Bnp) return;
    int b = t / npoint;
    const float* base = xyz + (size_t)b * bstride;
    float qx = nxyz[0 * Bnp + t], qy = nxyz[1 * Bnp + t], qz = nxyz[2 * Bnp + t];
    int* out = gidx + (size_t)t * nsample;
    int cnt = 0;
    for (int i = 0; i < n && cnt < nsample; ++i) {
        float dx = base[0 * cstride + i] - qx;
        float dy = base[1 * cstride + i] - qy;
        float dz = base[2 * cstride + i] - qz;
        if (dx * dx + dy * dy + dz * dz <= r2) out[cnt++] = i;
    }
    int f = (cnt > 0) ? out[0] : 0;
    for (; cnt < nsample; ++cnt) out[cnt] = f;
}

// ----------------------------- grouping -------------------------------------
// X ((3+Cp) x Cols), col = (b*npoint+m)*nsample + s.
__global__ void group_kernel(const float* __restrict__ xyz, int xb, int xc,
                             const float* __restrict__ pts, int pb, int pc, int Cp,
                             const float* __restrict__ nxyz, const int* __restrict__ gidx,
                             int Bnp, int npoint, int nsample,
                             float* __restrict__ X, int Cols)
{
    int col = blockIdx.x * blockDim.x + threadIdx.x;
    if (col >= Cols) return;
    int t = col / nsample;
    int s = col - t * nsample;
    int b = t / npoint;
    int g = gidx[(size_t)t * nsample + s];
    const float* xbase = xyz + (size_t)b * xb;
    X[(size_t)0 * Cols + col] = xbase[0 * xc + g] - nxyz[0 * Bnp + t];
    X[(size_t)1 * Cols + col] = xbase[1 * xc + g] - nxyz[1 * Bnp + t];
    X[(size_t)2 * Cols + col] = xbase[2 * xc + g] - nxyz[2 * Bnp + t];
    const float* pbase = pts + (size_t)b * pb;
    for (int c = 0; c < Cp; ++c)
        X[(size_t)(3 + c) * Cols + col] = pbase[(size_t)c * pc + g];
}

// ----------------------------- max pool over nsample -------------------------
__global__ void maxpool_kernel(const float* __restrict__ H, int Cols, int nsample,
                               float* __restrict__ Y, int Bnp, int C)
{
    int t = blockIdx.x * blockDim.x + threadIdx.x;
    if (t >= C * Bnp) return;
    int c = t / Bnp; int col = t - c * Bnp;
    const float* h = H + (size_t)c * Cols + (size_t)col * nsample;
    float m = h[0];
    for (int s = 1; s < nsample; ++s) m = fmaxf(m, h[s]);
    Y[(size_t)c * Bnp + col] = m;
}

// ----------------------------- 3-NN + weights --------------------------------
__global__ void three_nn_kernel(const float* __restrict__ dxyz, int db, int dc,
                                int ndst, int Bnd,
                                const float* __restrict__ sxyz, int sb, int sc,
                                int nsrc, int* __restrict__ idx3,
                                float* __restrict__ w3)
{
    int t = blockIdx.x * blockDim.x + threadIdx.x;
    if (t >= Bnd) return;
    int b = t / ndst; int m = t - b * ndst;
    const float* dbase = dxyz + (size_t)b * db;
    float qx = dbase[0 * dc + m], qy = dbase[1 * dc + m], qz = dbase[2 * dc + m];
    const float* sbase = sxyz + (size_t)b * sb;
    float d0 = 1e30f, d1 = 1e30f, d2 = 1e30f;
    int   i0 = 0,     i1 = 0,     i2 = 0;
    for (int i = 0; i < nsrc; ++i) {
        float dx = sbase[0 * sc + i] - qx;
        float dy = sbase[1 * sc + i] - qy;
        float dz = sbase[2 * sc + i] - qz;
        float d = dx * dx + dy * dy + dz * dz;
        if (d < d0)      { d2 = d1; i2 = i1; d1 = d0; i1 = i0; d0 = d; i0 = i; }
        else if (d < d1) { d2 = d1; i2 = i1; d1 = d;  i1 = i; }
        else if (d < d2) { d2 = d;  i2 = i; }
    }
    float r0 = 1.0f / (d0 + 1e-8f);
    float r1 = 1.0f / (d1 + 1e-8f);
    float r2 = 1.0f / (d2 + 1e-8f);
    float s = r0 + r1 + r2;
    idx3[t * 3 + 0] = i0; idx3[t * 3 + 1] = i1; idx3[t * 3 + 2] = i2;
    w3[t * 3 + 0] = r0 / s; w3[t * 3 + 1] = r1 / s; w3[t * 3 + 2] = r2 / s;
}

// ----------------------------- weighted interpolation ------------------------
__global__ void interp_kernel(const float* __restrict__ src, int srcColStride,
                              int nsrc, const int* __restrict__ idx3,
                              const float* __restrict__ w3,
                              float* __restrict__ dst, int Cols, int ndst,
                              int C, int rowoff)
{
    long long t = (long long)blockIdx.x * blockDim.x + threadIdx.x;
    long long total = (long long)C * Cols;
    if (t >= total) return;
    int c = (int)(t / Cols); int col = (int)(t - (long long)c * Cols);
    int b = col / ndst;
    const float* srow = src + (size_t)c * srcColStride + (size_t)b * nsrc;
    const int*   id = idx3 + (size_t)col * 3;
    const float* w  = w3   + (size_t)col * 3;
    dst[(size_t)(rowoff + c) * Cols + col] =
        srow[id[0]] * w[0] + srow[id[1]] * w[1] + srow[id[2]] * w[2];
}

// ----------------------------- FP3 broadcast of l3 ---------------------------
__global__ void bcast_l3_kernel(const float* __restrict__ P3, float* __restrict__ X,
                                int Cols, int npoint, int C, int rowoff)
{
    int t = blockIdx.x * blockDim.x + threadIdx.x;
    if (t >= C * Cols) return;
    int c = t / Cols; int col = t - c * Cols;
    int b = col / npoint;
    X[(size_t)(rowoff + c) * Cols + col] = P3[c * BATCH + b];
}

// ----------------------------- output writers --------------------------------
__global__ void write_cls_kernel(const float* __restrict__ P3, float* __restrict__ out)
{
    int t = blockIdx.x * blockDim.x + threadIdx.x;
    if (t >= BATCH * 1024) return;
    int b = t / 1024; int c = t - b * 1024;
    out[t] = P3[c * BATCH + b];             // P3: (1024 x B) -> (B,1024,1)
}
__global__ void write_seg_kernel(const float* __restrict__ S, float* __restrict__ out)
{
    long long t = (long long)blockIdx.x * blockDim.x + threadIdx.x;
    if (t >= (long long)BATCH * 128 * NPTS) return;
    int b = (int)(t / (128 * NPTS));
    int r = (int)(t % (128 * NPTS));
    int c = r / NPTS; int n = r - c * NPTS;
    out[t] = S[(size_t)c * (BATCH * NPTS) + (size_t)b * NPTS + n];
}

// ============================ host orchestration =============================
struct LayerP { const float *w, *b, *g, *beta, *m, *v; };

extern "C" void kernel_launch(void* const* d_in, const int* in_sizes, int n_in,
                              void* d_out, int out_size, void* d_ws, size_t ws_size,
                              hipStream_t stream)
{
    (void)out_size; (void)ws_size;
    if (n_in < 97) return;

    static const int LCI[16] = {12,64,64, 131,128,128, 259,256,512, 1280,256, 384,256, 128,128,128};
    static const int LCO[16] = {64,64,128, 128,128,256, 256,512,1024, 256,256, 256,128, 128,128,128};
    static const int LKP[16] = {32,64,64, 160,128,128, 288,256,512, 1280,256, 384,256, 128,128,128};

    LayerP L[16];
    const float* xyz;
    if (in_sizes[0] == BATCH * CINR * NPTS) {
        // insertion order: xyz first, then per layer: w,b,g,beta,m,v
        xyz = (const float*)d_in[0];
        int p = 1;
        for (int l = 0; l < 16; ++l) {
            L[l].w    = (const float*)d_in[p + 0];
            L[l].b    = (const float*)d_in[p + 1];
            L[l].g    = (const float*)d_in[p + 2];
            L[l].beta = (const float*)d_in[p + 3];
            L[l].m    = (const float*)d_in[p + 4];
            L[l].v    = (const float*)d_in[p + 5];
            p += 6;
        }
    } else {
        // jax-tree sorted keys: fp1_*, fp2_*, fp3_*, sa1_*, sa2_*, sa3_* ;
        // within layer: b, beta, g, m, v, w ; 'xyz' last.
        static const int ord[16] = {13,14,15, 11,12, 9,10, 0,1,2, 3,4,5, 6,7,8};
        int p = 0;
        for (int s = 0; s < 16; ++s) {
            int l = ord[s];
            L[l].b    = (const float*)d_in[p + 0];
            L[l].beta = (const float*)d_in[p + 1];
            L[l].g    = (const float*)d_in[p + 2];
            L[l].m    = (const float*)d_in[p + 3];
            L[l].v    = (const float*)d_in[p + 4];
            L[l].w    = (const float*)d_in[p + 5];
            p += 6;
        }
        xyz = (const float*)d_in[n_in - 1];
    }

    // -------- workspace carving (bump allocator, 256B aligned) --------
    uint8_t* ws = (uint8_t*)d_ws;
    size_t off = 0;
    auto alloc = [&](size_t bytes) -> void* {
        void* p = ws + off;
        off = (off + bytes + 255) & ~(size_t)255;
        return p;
    };
    const int BNP1 = BATCH * 512, BNP2 = BATCH * 128, BND0 = BATCH * NPTS;
    const int COLS1 = BNP1 * 32, COLS2 = BNP2 * 64;

    float*     nxyz1 = (float*)alloc((size_t)3 * BNP1 * 4);
    float*     nxyz2 = (float*)alloc((size_t)3 * BNP2 * 4);
    float*     P1    = (float*)alloc((size_t)128 * BNP1 * 4);   // l1_points
    float*     P2    = (float*)alloc((size_t)256 * BNP2 * 4);   // l2_points
    float*     P3    = (float*)alloc((size_t)1024 * BATCH * 4); // l3_points
    float*     P2f   = (float*)alloc((size_t)256 * BNP2 * 4);   // fp3 out
    float*     P1f   = (float*)alloc((size_t)128 * BNP1 * 4);   // fp2 out
    _Float16*  Wh    = (_Float16*)alloc((size_t)1024 * 1280 * 2); // padded f16 W
    int*   fidx1 = (int*)  alloc((size_t)BNP1 * 4);
    int*   fidx2 = (int*)  alloc((size_t)BNP2 * 4);
    int*   gidx1 = (int*)  alloc((size_t)COLS1 * 4);
    int*   gidx2 = (int*)  alloc((size_t)COLS2 * 4);
    int*   idx3a = (int*)  alloc((size_t)BNP1 * 3 * 4);
    float* w3a   = (float*)alloc((size_t)BNP1 * 3 * 4);
    int*   idx3b = (int*)  alloc((size_t)BND0 * 3 * 4);
    float* w3b   = (float*)alloc((size_t)BND0 * 3 * 4);
    float* R0 = (float*)alloc((size_t)160 * COLS2 * 4);     // max Kp x Cols X (~84MB)
    float* R1 = (float*)alloc((size_t)128 * COLS1 * 4);     // ping (~134MB)
    float* R2 = (float*)alloc((size_t)64  * COLS1 * 4);     // pong (~67MB)

    auto cdiv = [](long long a, long long b) { return (int)((a + b - 1) / b); };
    auto GEMM = [&](int li, const float* Xp, float* Yp, int Cols) {
        wcvt_kernel<<<cdiv((long long)LCO[li] * LKP[li], 256), 256, 0, stream>>>(
            L[li].w, Wh, LCO[li], LCI[li], LKP[li]);
        dim3 grid(Cols / 128, LCO[li] / 32);
        gemm_bn_relu<<<grid, 256, 0, stream>>>(Wh, Xp, Yp, LCO[li], LKP[li],
            Cols, L[li].b, L[li].g, L[li].beta, L[li].m, L[li].v, 1);
    };
    auto PADX = [&](int li, float* Xp, int Cols) {   // zero rows [K, Kp)
        long long n = (long long)(LKP[li] - LCI[li]) * Cols;
        if (n > 0)
            fill0_kernel<<<cdiv(n, 256), 256, 0, stream>>>(
                Xp + (size_t)LCI[li] * Cols, n);
    };

    // ================= SA1: 4096 -> 512 pts, r=0.2, ns=32, [64,64,128] ======
    fps_kernel<<<BATCH, 256, 0, stream>>>(xyz, CINR * NPTS, NPTS, NPTS, 512,
                                          fidx1, nxyz1, BNP1);
    query_ball_kernel<<<cdiv(BNP1, 256), 256, 0, stream>>>(
        xyz, CINR * NPTS, NPTS, NPTS, nxyz1, BNP1, 512, 32, 0.04f, gidx1);
    group_kernel<<<cdiv(COLS1, 256), 256, 0, stream>>>(
        xyz, CINR * NPTS, NPTS, xyz, CINR * NPTS, NPTS, CINR,
        nxyz1, gidx1, BNP1, 512, 32, R0, COLS1);
    PADX(0, R0, COLS1);
    GEMM(0, R0, R1, COLS1);
    GEMM(1, R1, R2, COLS1);
    GEMM(2, R2, R1, COLS1);
    maxpool_kernel<<<cdiv((long long)128 * BNP1, 256), 256, 0, stream>>>(
        R1, COLS1, 32, P1, BNP1, 128);

    // ================= SA2: 512 -> 128 pts, r=0.4, ns=64, [128,128,256] =====
    fps_kernel<<<BATCH, 256, 0, stream>>>(nxyz1, 512, BNP1, 512, 128,
                                          fidx2, nxyz2, BNP2);
    query_ball_kernel<<<cdiv(BNP2, 256), 256, 0, stream>>>(
        nxyz1, 512, BNP1, 512, nxyz2, BNP2, 128, 64, 0.16f, gidx2);
    group_kernel<<<cdiv(COLS2, 256), 256, 0, stream>>>(
        nxyz1, 512, BNP1, P1, 512, BNP1, 128,
        nxyz2, gidx2, BNP2, 128, 64, R0, COLS2);
    PADX(3, R0, COLS2);
    GEMM(3, R0, R1, COLS2);
    GEMM(4, R1, R2, COLS2);
    GEMM(5, R2, R1, COLS2);
    maxpool_kernel<<<cdiv((long long)256 * BNP2, 256), 256, 0, stream>>>(
        R1, COLS2, 64, P2, BNP2, 256);

    // ================= SA3: group_all over 128 pts, [256,512,1024] ==========
    // X3 (288 x 2048): rows 0-2 l2_xyz (absolute), 3-258 l2_points, pad to 288.
    hipMemcpyAsync(R0, nxyz2, (size_t)3 * BNP2 * 4, hipMemcpyDeviceToDevice, stream);
    hipMemcpyAsync(R0 + (size_t)3 * BNP2, P2, (size_t)256 * BNP2 * 4,
                   hipMemcpyDeviceToDevice, stream);
    PADX(6, R0, BNP2);
    GEMM(6, R0, R1, BNP2);
    GEMM(7, R1, R2, BNP2);
    GEMM(8, R2, R1, BNP2);
    maxpool_kernel<<<cdiv((long long)1024 * BATCH, 256), 256, 0, stream>>>(
        R1, BNP2, 128, P3, BATCH, 1024);
    write_cls_kernel<<<cdiv(BATCH * 1024, 256), 256, 0, stream>>>(P3, (float*)d_out);

    // ================= FP3: broadcast l3 -> l2, concat, [256,256] ===========
    hipMemcpyAsync(R0, P2, (size_t)256 * BNP2 * 4, hipMemcpyDeviceToDevice, stream);
    bcast_l3_kernel<<<cdiv(1024 * BNP2, 256), 256, 0, stream>>>(P3, R0, BNP2, 128,
                                                                1024, 256);
    GEMM(9,  R0, R1, BNP2);
    GEMM(10, R1, P2f, BNP2);

    // ================= FP2: 3-NN l2 -> l1, concat, [256,128] ================
    three_nn_kernel<<<cdiv(BNP1, 256), 256, 0, stream>>>(
        nxyz1, 512, BNP1, 512, BNP1, nxyz2, 128, BNP2, 128, idx3a, w3a);
    hipMemcpyAsync(R0, P1, (size_t)128 * BNP1 * 4, hipMemcpyDeviceToDevice, stream);
    interp_kernel<<<cdiv((long long)256 * BNP1, 256), 256, 0, stream>>>(
        P2f, BNP2, 128, idx3a, w3a, R0, BNP1, 512, 256, 128);
    GEMM(11, R0, R1, BNP1);
    GEMM(12, R1, P1f, BNP1);

    // ================= FP1: 3-NN l1 -> l0, [128,128,128] ====================
    three_nn_kernel<<<cdiv(BND0, 256), 256, 0, stream>>>(
        xyz, CINR * NPTS, NPTS, NPTS, BND0, nxyz1, 512, BNP1, 512, idx3b, w3b);
    interp_kernel<<<cdiv((long long)128 * BND0, 256), 256, 0, stream>>>(
        P1f, BNP1, 512, idx3b, w3b, R0, BND0, NPTS, 128, 0);
    GEMM(13, R0, R1, BND0);
    GEMM(14, R1, R2, BND0);
    GEMM(15, R2, R1, BND0);
    write_seg_kernel<<<cdiv((long long)BATCH * 128 * NPTS, 256), 256, 0, stream>>>(
        R1, (float*)d_out + (size_t)BATCH * 1024);
}
